// PocketEncoder_73607149519587
// MI455X (gfx1250) — matrix operand order repaired and checked by reading, not compiled, not run on hardware
//
#include <hip/hip_runtime.h>

#define B_        2
#define N_        256
#define NC_       512
#define NEGV      (-1000000000.0f)

typedef __attribute__((ext_vector_type(16))) _Float16 v16h;
typedef __attribute__((ext_vector_type(4)))  _Float16 v4h;
typedef __attribute__((ext_vector_type(8)))  float    v8f;

union Frag16 { v16h v; uint4 q[2]; };

#define FSTR 272   // halves per row of feats/h tile in LDS (16B aligned, padded)
#define WSTR 272   // halves per column of transposed weights in LDS

__device__ __forceinline__ float fast_rcp(float x) { return __builtin_amdgcn_rcpf(x); }

// ---------------------------------------------------------------------------
// Kernel A: small dense projections (scalar; negligible work)
// ---------------------------------------------------------------------------
__global__ __launch_bounds__(256)
void precompute_kernel(const float* __restrict__ invs, const float* __restrict__ cond_invs,
                       const float* __restrict__ cond_equis,
                       const float* __restrict__ q_w, const float* __restrict__ q_b,
                       const float* __restrict__ k_w, const float* __restrict__ k_b,
                       const float* __restrict__ inv_in_w, const float* __restrict__ inv_in_b,
                       const float* __restrict__ equi_in_w,
                       float* __restrict__ qm, float* __restrict__ km,
                       float* __restrict__ hf, float* __restrict__ proj)
{
    int idx = blockIdx.x * 256 + threadIdx.x;
    const int n_qm = B_ * N_ * 64;
    const int n_km = B_ * NC_ * 64;
    const int n_hf = B_ * NC_ * 256;
    const int n_pj = B_ * NC_ * 3 * 64;
    if (idx < n_qm) {
        int row = idx >> 6, c = idx & 63;
        float a = q_b[c];
        const float* x = invs + (size_t)row * 256;
        for (int kk = 0; kk < 256; ++kk) a += x[kk] * q_w[kk * 64 + c];
        qm[idx] = a; return;
    }
    idx -= n_qm;
    if (idx < n_km) {
        int row = idx >> 6, c = idx & 63;
        float a = k_b[c];
        const float* x = cond_invs + (size_t)row * 256;
        for (int kk = 0; kk < 256; ++kk) a += x[kk] * k_w[kk * 64 + c];
        km[idx] = a; return;
    }
    idx -= n_km;
    if (idx < n_hf) {
        int row = idx >> 8, c = idx & 255;
        float a = inv_in_b[c];
        const float* x = cond_invs + (size_t)row * 256;
        for (int kk = 0; kk < 256; ++kk) a += x[kk] * inv_in_w[kk * 256 + c];
        hf[idx] = a; return;
    }
    idx -= n_hf;
    if (idx < n_pj) {
        int row = idx >> 6, e = idx & 63;
        float a = 0.f;
        const float* x = cond_equis + (size_t)row * 64;
        for (int dd = 0; dd < 64; ++dd) a += x[dd] * equi_in_w[dd * 64 + e];
        proj[idx] = a; return;
    }
}

// ---------------------------------------------------------------------------
// Kernel B: fused per-edge MLP via WMMA f16 (the 54-GFLOP core).
// Block = 256 threads = 8 waves; tile = 16 q x 16 k = 256 rows, K=256.
// Wave w owns rows [32w, 32w+32) = 2 M-tiles (q = qt*16 + 2w and +1).
// Each B fragment (shared w1/w2 column slice) feeds TWO WMMAs; B fragments
// are double-buffered so ds_load_b128 overlaps with v_wmma.
// ---------------------------------------------------------------------------
__global__ __launch_bounds__(256)
void fused_messages_wmma(const float* __restrict__ equis,
                         const float* __restrict__ cond_equis,
                         const float* __restrict__ edges,
                         const float* __restrict__ q_msg,
                         const float* __restrict__ k_msg,
                         const float* __restrict__ w1,
                         const float* __restrict__ b1,
                         const float* __restrict__ w2,
                         const float* __restrict__ b2,
                         float* __restrict__ logits,
                         float* __restrict__ edge_out)
{
    extern __shared__ _Float16 smem[];
    _Float16* s_feats = smem;                // [256][FSTR] feats tile, later h tile
    _Float16* s_w     = smem + 256 * FSTR;   // [256][WSTR] transposed weights: s_w[col*WSTR+k]

    const int tid = threadIdx.x;
    const int b   = blockIdx.z;
    const int qt  = blockIdx.y;
    const int kt  = blockIdx.x;

    // ---- Stage feats tile (256 rows x 256 cols) into LDS as f16, 4 cols/iter.
    for (int idx = tid; idx < 256 * 64; idx += 256) {
        int r  = idx >> 6;
        int c4 = (idx & 63) << 2;
        int qi = r >> 4, ki = r & 15;
        int q = qt * 16 + qi, k = kt * 16 + ki;
        float4 f;
        if (c4 < 64) {
            f = *(const float4*)(q_msg + ((size_t)b * N_ + q) * 64 + c4);
        } else if (c4 < 128) {
            f = *(const float4*)(k_msg + ((size_t)b * NC_ + k) * 64 + (c4 - 64));
        } else if (c4 < 192) {
            int d = c4 - 128;
            const float4* eq = (const float4*)(equis      + (((size_t)b * N_  + q) * 3) * 64 + d);
            const float4* ce = (const float4*)(cond_equis + (((size_t)b * NC_ + k) * 3) * 64 + d);
            float4 a0 = eq[0], a1 = eq[16], a2 = eq[32];
            float4 c0 = ce[0], c1 = ce[16], c2 = ce[32];
            f.x = a0.x * c0.x + a1.x * c1.x + a2.x * c2.x;
            f.y = a0.y * c0.y + a1.y * c1.y + a2.y * c2.y;
            f.z = a0.z * c0.z + a1.z * c1.z + a2.z * c2.z;
            f.w = a0.w * c0.w + a1.w * c1.w + a2.w * c2.w;
        } else {
            f = *(const float4*)(edges + (((size_t)b * N_ + q) * NC_ + k) * 64 + (c4 - 192));
        }
        v4h hv = { (_Float16)f.x, (_Float16)f.y, (_Float16)f.z, (_Float16)f.w };
        *(v4h*)(s_feats + r * FSTR + c4) = hv;
    }
    // ---- Stage w1 transposed into LDS (f16): s_w[col][k] = w1[k][col]
    for (int idx = tid; idx < 256 * 64; idx += 256) {
        int r = idx >> 6, c4 = (idx & 63) << 2;
        float4 wv = *(const float4*)(w1 + (size_t)r * 256 + c4);
        s_w[(c4 + 0) * WSTR + r] = (_Float16)wv.x;
        s_w[(c4 + 1) * WSTR + r] = (_Float16)wv.y;
        s_w[(c4 + 2) * WSTR + r] = (_Float16)wv.z;
        s_w[(c4 + 3) * WSTR + r] = (_Float16)wv.w;
    }
    __syncthreads();

    const int wave  = tid >> 5;
    const int lane  = tid & 31;
    const int mcol  = lane & 15;   // A: M row of lane; B/C: N column of lane
    const int half_ = lane >> 4;

    // ---- A fragments: two M-tiles per wave (rows 32w.. and 32w+16..)
    Frag16 af0[8], af1[8];
    {
        const _Float16* rp0 = s_feats + (wave * 32 + mcol) * FSTR;
        const _Float16* rp1 = rp0 + 16 * FSTR;
#pragma unroll
        for (int kc = 0; kc < 8; ++kc) {
            af0[kc].q[0] = *(const uint4*)(rp0 + kc * 32 + half_ * 8);
            af0[kc].q[1] = *(const uint4*)(rp0 + kc * 32 + 16 + half_ * 8);
            af1[kc].q[0] = *(const uint4*)(rp1 + kc * 32 + half_ * 8);
            af1[kc].q[1] = *(const uint4*)(rp1 + kc * 32 + 16 + half_ * 8);
        }
    }

    // ---- GEMM1: h = silu(feats @ w1 + b1); write h over the feats rows.
#pragma unroll 1
    for (int nt = 0; nt < 16; ++nt) {
        v8f acc0 = {}, acc1 = {};
        const int col = nt * 16 + mcol;
        const _Float16* wp = s_w + col * WSTR + half_ * 8;
        Frag16 bf[2];
        bf[0].q[0] = *(const uint4*)(wp);
        bf[0].q[1] = *(const uint4*)(wp + 16);
#pragma unroll
        for (int kc = 0; kc < 8; ++kc) {
            if (kc < 7) {
                bf[(kc + 1) & 1].q[0] = *(const uint4*)(wp + (kc + 1) * 32);
                bf[(kc + 1) & 1].q[1] = *(const uint4*)(wp + (kc + 1) * 32 + 16);
            }
            acc0 = __builtin_amdgcn_wmma_f32_16x16x32_f16(false, af0[kc].v, false, bf[kc & 1].v,
                                                          (short)0, acc0, false, false);
            acc1 = __builtin_amdgcn_wmma_f32_16x16x32_f16(false, af1[kc].v, false, bf[kc & 1].v,
                                                          (short)0, acc1, false, false);
        }
        float bias = b1[col];
#pragma unroll
        for (int r = 0; r < 8; ++r) {
            float x0 = acc0[r] + bias;
            float h0 = x0 * fast_rcp(1.0f + __expf(-x0));                  // silu
            s_feats[(wave * 32 + r + 8 * half_) * FSTR + col] = (_Float16)h0;
            float x1 = acc1[r] + bias;
            float h1 = x1 * fast_rcp(1.0f + __expf(-x1));
            s_feats[(wave * 32 + 16 + r + 8 * half_) * FSTR + col] = (_Float16)h1;
        }
    }
    __syncthreads();

    // ---- Swap weights: w2 (256 x 144) transposed into LDS.
    for (int idx = tid; idx < 256 * 36; idx += 256) {
        int r = idx / 36, c4 = (idx - r * 36) << 2;
        float4 wv = *(const float4*)(w2 + (size_t)r * 144 + c4);
        s_w[(c4 + 0) * WSTR + r] = (_Float16)wv.x;
        s_w[(c4 + 1) * WSTR + r] = (_Float16)wv.y;
        s_w[(c4 + 2) * WSTR + r] = (_Float16)wv.z;
        s_w[(c4 + 3) * WSTR + r] = (_Float16)wv.w;
    }
    __syncthreads();

    // ---- Reload A fragments (now h).
    {
        const _Float16* rp0 = s_feats + (wave * 32 + mcol) * FSTR;
        const _Float16* rp1 = rp0 + 16 * FSTR;
#pragma unroll
        for (int kc = 0; kc < 8; ++kc) {
            af0[kc].q[0] = *(const uint4*)(rp0 + kc * 32 + half_ * 8);
            af0[kc].q[1] = *(const uint4*)(rp0 + kc * 32 + 16 + half_ * 8);
            af1[kc].q[0] = *(const uint4*)(rp1 + kc * 32 + half_ * 8);
            af1[kc].q[1] = *(const uint4*)(rp1 + kc * 32 + 16 + half_ * 8);
        }
    }

    // ---- GEMM2: messages = h @ w2 + b2 (cols 0..143 over 9 N-tiles).
    const int q0 = qt * 16 + 2 * wave;
    const int q1 = q0 + 1;
#pragma unroll 1
    for (int nt = 0; nt < 9; ++nt) {
        v8f acc0 = {}, acc1 = {};
        const int col = nt * 16 + mcol;
        const _Float16* wp = s_w + col * WSTR + half_ * 8;
        Frag16 bf[2];
        bf[0].q[0] = *(const uint4*)(wp);
        bf[0].q[1] = *(const uint4*)(wp + 16);
#pragma unroll
        for (int kc = 0; kc < 8; ++kc) {
            if (kc < 7) {
                bf[(kc + 1) & 1].q[0] = *(const uint4*)(wp + (kc + 1) * 32);
                bf[(kc + 1) & 1].q[1] = *(const uint4*)(wp + (kc + 1) * 32 + 16);
            }
            acc0 = __builtin_amdgcn_wmma_f32_16x16x32_f16(false, af0[kc].v, false, bf[kc & 1].v,
                                                          (short)0, acc0, false, false);
            acc1 = __builtin_amdgcn_wmma_f32_16x16x32_f16(false, af1[kc].v, false, bf[kc & 1].v,
                                                          (short)0, acc1, false, false);
        }
        float bias = b2[col];
#pragma unroll
        for (int r = 0; r < 8; ++r) {
            int k = kt * 16 + r + 8 * half_;
            float v0 = acc0[r] + bias;
            float v1 = acc1[r] + bias;
            if (col < 80) {
                logits[(((size_t)b * N_ + q0) * NC_ + k) * 80 + col] = v0;
                logits[(((size_t)b * N_ + q1) * NC_ + k) * 80 + col] = v1;
            } else {
                edge_out[(((size_t)b * N_ + q0) * NC_ + k) * 64 + (col - 80)] = v0;
                edge_out[(((size_t)b * N_ + q1) * NC_ + k) * 64 + (col - 80)] = v1;
            }
        }
    }
}

// ---------------------------------------------------------------------------
// Kernel C1: equi path. One block per (b,q): masked softmax over k per d,
// fused attn*proj accumulation, w_e scaling, 64x64 output projection.
// ---------------------------------------------------------------------------
__global__ __launch_bounds__(256)
void equi_attn_kernel(const float* __restrict__ logits, const int* __restrict__ adj,
                      const float* __restrict__ proj, const float* __restrict__ equi_out_w,
                      float* __restrict__ out_equi)
{
    __shared__ float sred[4][64];
    __shared__ float seo[3][64];
    const int bq = blockIdx.x;          // b*N + q
    const int b  = bq / N_;
    const int t  = threadIdx.x;
    const int d  = t & 63;
    const int g  = t >> 6;              // 4 groups of 128 k's

    const float* lg = logits + (size_t)bq * NC_ * 80;
    const int*   am = adj    + (size_t)bq * NC_;
    const float* pj = proj   + ((size_t)b * NC_) * 3 * 64;

    float m = -3.4e38f;
    for (int k = g * 128; k < g * 128 + 128; ++k) {
        float v = lg[(size_t)k * 80 + d] + (am[k] > 0 ? 0.0f : NEGV);
        m = fmaxf(m, v);
    }
    sred[g][d] = m;
    __syncthreads();
    m = fmaxf(fmaxf(sred[0][d], sred[1][d]), fmaxf(sred[2][d], sred[3][d]));
    __syncthreads();

    float s = 0.f, sq = 0.f, e0 = 0.f, e1 = 0.f, e2 = 0.f;
    for (int k = g * 128; k < g * 128 + 128; ++k) {
        float v = lg[(size_t)k * 80 + d] + (am[k] > 0 ? 0.0f : NEGV);
        float e = __expf(v - m);
        s += e; sq += e * e;
        const float* pr = pj + (size_t)k * 3 * 64 + d;
        e0 += e * pr[0]; e1 += e * pr[64]; e2 += e * pr[128];
    }
    float vals[5] = {s, sq, e0, e1, e2};
#pragma unroll
    for (int i = 0; i < 5; ++i) {
        sred[g][d] = vals[i];
        __syncthreads();
        vals[i] = sred[0][d] + sred[1][d] + sred[2][d] + sred[3][d];
        __syncthreads();
    }
    float inv_s = fast_rcp(vals[0]);
    float we    = sqrtf(vals[1]) * inv_s;   // sqrt(sum attn^2)
    float scale = inv_s * we;
    if (g == 0) {
        seo[0][d] = vals[2] * scale;
        seo[1][d] = vals[3] * scale;
        seo[2][d] = vals[4] * scale;
    }
    __syncthreads();
    if (t < 192) {
        int x = t >> 6, e = t & 63;
        float acc = 0.f;
        for (int dd = 0; dd < 64; ++dd) acc += seo[x][dd] * equi_out_w[dd * 64 + e];
        out_equi[((size_t)bq * 3 + x) * 64 + e] = acc;
    }
}

// ---------------------------------------------------------------------------
// Kernel C2: inv path. One block per (b,q): 16-head masked softmax (attn
// cached in LDS), attn*hf accumulation, w_i scaling, 256x256 out projection.
// ---------------------------------------------------------------------------
__global__ __launch_bounds__(256)
void inv_attn_kernel(const float* __restrict__ logits, const int* __restrict__ adj,
                     const float* __restrict__ hf, const float* __restrict__ inv_out_w,
                     const float* __restrict__ inv_out_b, float* __restrict__ out_inv)
{
    __shared__ float sattn[NC_ * 16];   // 32 KB
    __shared__ float sred[16][16];
    __shared__ float sfac[16];
    __shared__ float sio[256];
    const int bq = blockIdx.x;
    const int b  = bq / N_;
    const int t  = threadIdx.x;
    const int h  = t & 15;
    const int g  = t >> 4;              // 16 groups of 32 k's

    const float* lg = logits + (size_t)bq * NC_ * 80 + 64;   // inv logits cols 64..79
    const int*   am = adj    + (size_t)bq * NC_;

    float m = -3.4e38f;
    for (int k = g * 32; k < g * 32 + 32; ++k) {
        float v = lg[(size_t)k * 80 + h] + (am[k] > 0 ? 0.0f : NEGV);
        m = fmaxf(m, v);
    }
    sred[g][h] = m;
    __syncthreads();
    m = -3.4e38f;
#pragma unroll
    for (int i = 0; i < 16; ++i) m = fmaxf(m, sred[i][h]);
    __syncthreads();

    float s = 0.f, sq = 0.f;
    for (int k = g * 32; k < g * 32 + 32; ++k) {
        float v = lg[(size_t)k * 80 + h] + (am[k] > 0 ? 0.0f : NEGV);
        float e = __expf(v - m);
        sattn[k * 16 + h] = e;
        s += e; sq += e * e;
    }
    sred[g][h] = s;  __syncthreads();
    s = 0.f;
#pragma unroll
    for (int i = 0; i < 16; ++i) s += sred[i][h];
    __syncthreads();
    sred[g][h] = sq; __syncthreads();
    sq = 0.f;
#pragma unroll
    for (int i = 0; i < 16; ++i) sq += sred[i][h];
    __syncthreads();
    if (g == 0) sfac[h] = sqrtf(sq) * fast_rcp(s * s);   // w_i / s (attn unnormalized)
    __syncthreads();

    {   // io[h][dd] = (sum_k e * hf) * sfac
        int h2 = t >> 4, dd = t & 15;
        const float* hp = hf + ((size_t)b * NC_) * 256 + h2 * 16 + dd;
        float acc = 0.f;
        for (int k = 0; k < NC_; ++k) acc += sattn[k * 16 + h2] * hp[(size_t)k * 256];
        sio[t] = acc * sfac[h2];
    }
    __syncthreads();
    {   // inv_out = io @ inv_out_w + b
        float o = inv_out_b[t];
        for (int dct = 0; dct < 256; ++dct) o += sio[dct] * inv_out_w[dct * 256 + t];
        out_inv[(size_t)bq * 256 + t] = o;
    }
}

// ---------------------------------------------------------------------------
extern "C" void kernel_launch(void* const* d_in, const int* in_sizes, int n_in,
                              void* d_out, int out_size, void* d_ws, size_t ws_size,
                              hipStream_t stream)
{
    const float* equis      = (const float*)d_in[0];
    const float* invs       = (const float*)d_in[1];
    const float* cond_equis = (const float*)d_in[2];
    const float* cond_invs  = (const float*)d_in[3];
    const float* edges      = (const float*)d_in[4];
    const int*   adj        = (const int*)  d_in[5];
    const float* q_w        = (const float*)d_in[6];
    const float* q_b        = (const float*)d_in[7];
    const float* k_w        = (const float*)d_in[8];
    const float* k_b        = (const float*)d_in[9];
    const float* w1         = (const float*)d_in[10];
    const float* b1         = (const float*)d_in[11];
    const float* w2         = (const float*)d_in[12];
    const float* b2         = (const float*)d_in[13];
    const float* equi_in_w  = (const float*)d_in[14];
    const float* equi_out_w = (const float*)d_in[15];
    const float* inv_in_w   = (const float*)d_in[16];
    const float* inv_in_b   = (const float*)d_in[17];
    const float* inv_out_w  = (const float*)d_in[18];
    const float* inv_out_b  = (const float*)d_in[19];

    float* out      = (float*)d_out;
    float* out_equi = out;                                   // B*N*3*64   = 98304
    float* out_inv  = out + (size_t)B_ * N_ * 3 * 64;        // B*N*256    = 131072
    float* out_edge = out_inv + (size_t)B_ * N_ * 256;       // B*N*NC*64  = 16777216

    float* ws     = (float*)d_ws;
    float* logits = ws;                                      // B*N*NC*80 fp32 (~84 MB)
    float* qm     = logits + (size_t)B_ * N_ * NC_ * 80;
    float* km     = qm + (size_t)B_ * N_ * 64;
    float* hf     = km + (size_t)B_ * NC_ * 64;
    float* proj   = hf + (size_t)B_ * NC_ * 256;

    // A: small projections
    {
        int totalA = B_ * N_ * 64 + B_ * NC_ * 64 + B_ * NC_ * 256 + B_ * NC_ * 3 * 64;
        precompute_kernel<<<(totalA + 255) / 256, 256, 0, stream>>>(
            invs, cond_invs, cond_equis, q_w, q_b, k_w, k_b,
            inv_in_w, inv_in_b, equi_in_w, qm, km, hf, proj);
    }

    // B: fused per-edge MLP (WMMA)
    {
        dim3 gB(NC_ / 16, N_ / 16, B_);   // 32 x 16 x 2 = 1024 blocks
        size_t ldsB = (size_t)(256 * FSTR + 256 * WSTR) * sizeof(_Float16);  // ~272 KB
        fused_messages_wmma<<<gB, 256, ldsB, stream>>>(
            equis, cond_equis, edges, qm, km, w1, b1, w2, b2, logits, out_edge);
    }

    // C1 / C2: attention reductions + output projections
    equi_attn_kernel<<<B_ * N_, 256, 0, stream>>>(logits, adj, proj, equi_out_w, out_equi);
    inv_attn_kernel<<<B_ * N_, 256, 0, stream>>>(logits, adj, hf, inv_out_w, inv_out_b, out_inv);
}
